// MinGRUParallel_82162724372987
// MI455X (gfx1250) — matrix operand order repaired and checked by reading, not compiled, not run on hardware
//
#include <hip/hip_runtime.h>
#include <hip/hip_bf16.h>

typedef __attribute__((ext_vector_type(16))) __bf16 v16bf;
typedef __attribute__((ext_vector_type(8)))  float  v8f;

#define B_  32
#define T_  1024
#define I_  256
#define H_  512
#define WC_ (I_ + H_)      // 768 columns in Wz/Wh
#define BT_ (B_ * T_)      // 32768 rows

// ---- float -> bf16 with round-to-nearest-even (used only OUTSIDE hot loops) ----
__device__ __forceinline__ __bf16 f2bf(float f) {
    union { float f; unsigned u; } v; v.f = f;
    unsigned r = v.u + 0x7FFFu + ((v.u >> 16) & 1u);
    unsigned short h = (unsigned short)(r >> 16);
    __bf16 o; __builtin_memcpy(&o, &h, 2); return o;
}

// A fragment (16x32 bf16, ISA 7.12.2): lane row = lane&15, khalf = lane>>4.
// element e -> K = k0 + khalf*16 + e (matches B fragment order)
// bf16 source: two 16-byte loads, zero conversion ALU.
__device__ __forceinline__ v16bf load_a_frag_bf(const __bf16* __restrict__ row,
                                                int khalf, int k0) {
    union { v16bf v; uint4 q[2]; } u;
    u.q[0] = *(const uint4*)(row + k0 + khalf * 16);
    u.q[1] = *(const uint4*)(row + k0 + khalf * 16 + 8);
    return u.v;
}

// B fragment (32x16 bf16): lane col = lane&15, element e -> K = kbase + (lane>>4)*16 + e
// 16 contiguous bf16 per lane -> two 16-byte loads.
__device__ __forceinline__ v16bf load_b_frag_bf(const __bf16* __restrict__ p) {
    union { v16bf v; uint4 q[2]; } u;
    u.q[0] = *(const uint4*)(p);
    u.q[1] = *(const uint4*)(p + 8);
    return u.v;
}

// ---- one-time pass: bf16 conversions of x/Wz/Wh, zero h state + barrier counters ----
__global__ __launch_bounds__(256) void mingru_prep(
    const float* __restrict__ x, const float* __restrict__ Wz, const float* __restrict__ Wh,
    __bf16* __restrict__ xbf, __bf16* __restrict__ wzbf, __bf16* __restrict__ whbf,
    float* __restrict__ hb0, float* __restrict__ hb1,
    __bf16* __restrict__ hbf0, __bf16* __restrict__ hbf1,
    unsigned* __restrict__ cnt)
{
    int i = blockIdx.x * 256 + threadIdx.x;
    int stride = gridDim.x * 256;
    for (int k = i; k < BT_ * I_; k += stride) xbf[k] = f2bf(x[k]);
    for (int k = i; k < H_ * WC_; k += stride) { wzbf[k] = f2bf(Wz[k]); whbf[k] = f2bf(Wh[k]); }
    for (int k = i; k < B_ * H_; k += stride) {
        hb0[k] = 0.0f; hb1[k] = 0.0f;
        hbf0[k] = (__bf16)0.0f; hbf1[k] = (__bf16)0.0f;
    }
    for (int k = i; k < T_; k += stride) cnt[k] = 0u;
}

// ---- projection GEMM: xz = x@WzxT + bz (-> ws), xh = x@WhxT + bh (-> d_out) ----
// one wave per 16x16 output tile; K = 256 in 8 WMMA steps of 32; pure bf16 loads
__global__ __launch_bounds__(256) void mingru_proj(
    const __bf16* __restrict__ xbf,
    const __bf16* __restrict__ wzbf, const float* __restrict__ bz,
    const __bf16* __restrict__ whbf, const float* __restrict__ bh,
    float* __restrict__ xz, float* __restrict__ xh)
{
    int lane  = threadIdx.x & 31;
    int wid   = blockIdx.x * 8 + (threadIdx.x >> 5);
    int gate  = wid >> 16;               // 65536 tiles per gate
    int rem   = wid & 0xFFFF;
    int rt    = rem >> 5;                // 2048 row tiles
    int ct    = rem & 31;                // 32 col tiles
    int r0 = rt * 16, c0 = ct * 16;
    const __bf16* W    = gate ? whbf : wzbf;
    const float*  bias = gate ? bh : bz;
    float*        out  = gate ? xh : xz;

    int nl = lane & 15, khalf = lane >> 4;
    const __bf16* arow = xbf + (size_t)(r0 + nl) * I_;
    const __bf16* wrow = W   + (size_t)(c0 + nl) * WC_;   // x-part: cols [0, I)

    v8f acc = {};
    #pragma unroll
    for (int k0 = 0; k0 < I_; k0 += 32) {
        v16bf a = load_a_frag_bf(arow, khalf, k0);
        v16bf b = load_b_frag_bf(wrow + k0 + khalf * 16);
        acc = __builtin_amdgcn_wmma_f32_16x16x32_bf16(false, a, false, b,
                                                      (short)0, acc, false, false);
    }
    float bv = bias[c0 + nl];
    #pragma unroll
    for (int j = 0; j < 8; ++j) {
        int m = j + 8 * khalf;           // D: element j -> row j + 8*(lane>>4)
        out[(size_t)(r0 + m) * H_ + c0 + nl] = acc[j] + bv;
    }
}

// ---- sequential recurrence: 32 WGs, WG g owns H cols [16g,16g+16) for both gates.
// Recurrent-weight B fragments live in VGPRs. Hidden state kept in fp32 (carried)
// + bf16 mirror (matmul A input). 4 waves/WG: wave = (gate<<1)|mtile.
__global__ __launch_bounds__(128) void mingru_recur(
    const __bf16* __restrict__ wzbf, const __bf16* __restrict__ whbf,
    const float* __restrict__ xz,
    float* __restrict__ out,                       // holds xh, overwritten with h_t
    float* __restrict__ hb0, float* __restrict__ hb1,
    __bf16* __restrict__ hbf0, __bf16* __restrict__ hbf1,
    unsigned* __restrict__ cnt)
{
    __shared__ float htl[2 * 16 * 16];             // h_tilde exchange (2 mtiles)

    int lane = threadIdx.x & 31;
    int w    = threadIdx.x >> 5;
    int gt   = w >> 1;                             // 0 = z-gate, 1 = h-gate
    int mt   = w & 1;                              // batch tile (rows 0-15 / 16-31)
    int c0   = blockIdx.x * 16;
    int nl = lane & 15, khalf = lane >> 4;

    // preload this wave's recurrent-weight B fragments (h-part: cols [I, I+H))
    const __bf16* Wrow = (gt ? whbf : wzbf) + (size_t)(c0 + nl) * WC_ + I_;
    v16bf bf[16];
    #pragma unroll
    for (int kk = 0; kk < 16; ++kk)
        bf[kk] = load_b_frag_bf(Wrow + kk * 32 + khalf * 16);

    int arow_b = mt * 16 + nl;                     // A-matrix row (batch index)
    int col    = c0 + nl;

    for (int t = 0; t < T_; ++t) {
        const float*  hp    = (t & 1) ? hb1  : hb0;
        float*        hn    = (t & 1) ? hb0  : hb1;
        const __bf16* hp_bf = (t & 1) ? hbf1 : hbf0;
        __bf16*       hn_bf = (t & 1) ? hbf0 : hbf1;

        const __bf16* hrow = hp_bf + (size_t)arow_b * H_;
        v8f acc = {};
        #pragma unroll
        for (int kk = 0; kk < 16; ++kk) {
            v16bf a = load_a_frag_bf(hrow, khalf, kk * 32);
            acc = __builtin_amdgcn_wmma_f32_16x16x32_bf16(false, a, false, bf[kk],
                                                          (short)0, acc, false, false);
        }

        if (gt) {   // candidate gate: tanh(xh_t + h@WhhT) -> LDS
            #pragma unroll
            for (int j = 0; j < 8; ++j) {
                int m  = j + 8 * khalf;
                int bj = mt * 16 + m;
                float pre = acc[j] + out[((size_t)bj * T_ + t) * H_ + col];
                float th  = 1.0f - 2.0f / (__expf(2.0f * pre) + 1.0f);
                htl[(mt * 16 + m) * 16 + nl] = th;
            }
        }
        __syncthreads();
        if (!gt) {  // update gate: blend, write h_new (fp32 + bf16 mirror) + output
            #pragma unroll
            for (int j = 0; j < 8; ++j) {
                int m  = j + 8 * khalf;
                int bj = mt * 16 + m;
                float pre   = acc[j] + xz[((size_t)bj * T_ + t) * H_ + col];
                float z     = 1.0f / (1.0f + __expf(-pre));
                float hprev = hp[(size_t)bj * H_ + col];
                float ht    = htl[(mt * 16 + m) * 16 + nl];
                float hnew  = hprev + z * (ht - hprev);
                hn[(size_t)bj * H_ + col]    = hnew;
                hn_bf[(size_t)bj * H_ + col] = f2bf(hnew);
                out[((size_t)bj * T_ + t) * H_ + col] = hnew;
            }
        }
        __syncthreads();
        // device-scope step barrier across the 32 workgroups (release/acquire)
        if (threadIdx.x == 0) {
            __hip_atomic_fetch_add(&cnt[t], 1u, __ATOMIC_RELEASE, __HIP_MEMORY_SCOPE_AGENT);
            while (__hip_atomic_load(&cnt[t], __ATOMIC_ACQUIRE, __HIP_MEMORY_SCOPE_AGENT) < 32u)
                __builtin_amdgcn_s_sleep(2);
        }
        __syncthreads();
    }
}

extern "C" void kernel_launch(void* const* d_in, const int* in_sizes, int n_in,
                              void* d_out, int out_size, void* d_ws, size_t ws_size,
                              hipStream_t stream) {
    (void)in_sizes; (void)n_in; (void)out_size; (void)ws_size;
    const float* x  = (const float*)d_in[0];
    const float* Wz = (const float*)d_in[1];
    const float* bz = (const float*)d_in[2];
    const float* Wh = (const float*)d_in[3];
    const float* bh = (const float*)d_in[4];
    float* out = (float*)d_out;

    // ---- workspace layout (fp32 region first, then bf16 region) ----
    char* ws = (char*)d_ws;
    float* xz  = (float*)ws;                                  // BT*H f32   (64 MB)
    float* hb0 = xz + (size_t)BT_ * H_;                       // B*H  f32
    float* hb1 = hb0 + B_ * H_;                               // B*H  f32
    unsigned* cnt = (unsigned*)(hb1 + B_ * H_);               // T counters
    __bf16* xbf  = (__bf16*)(cnt + T_);                       // BT*I bf16  (16 MB)
    __bf16* wzbf = xbf + (size_t)BT_ * I_;                    // H*WC bf16
    __bf16* whbf = wzbf + (size_t)H_ * WC_;                   // H*WC bf16
    __bf16* hbf0 = whbf + (size_t)H_ * WC_;                   // B*H  bf16
    __bf16* hbf1 = hbf0 + B_ * H_;                            // B*H  bf16

    mingru_prep <<<2048, 256, 0, stream>>>(x, Wz, Wh, xbf, wzbf, whbf,
                                           hb0, hb1, hbf0, hbf1, cnt);
    mingru_proj <<<16384, 256, 0, stream>>>(xbf, wzbf, bz, whbf, bh, xz, out);
    mingru_recur<<<32, 128, 0, stream>>>(wzbf, whbf, xz, out,
                                         hb0, hb1, hbf0, hbf1, cnt);
}
